// GConvGRUv2_14336600834602
// MI455X (gfx1250) — matrix operand order repaired and checked by reading, not compiled, not run on hardware
//
#include <hip/hip_runtime.h>
#include <hip/hip_bf16.h>
#include <math.h>

// ---------------------------------------------------------------------------
// GConvGRU (ChebConv K=3, C=1024, N=20000, E=160000) for gfx1250.
// GEMMs (755 GFLOP dominant) run on v_wmma_f32_16x16x32_bf16 with f32 accum.
// ---------------------------------------------------------------------------

#define N_NODES 20000
#define N_EDGES 160000
#define CH 1024
#define NCH ((size_t)N_NODES * CH)   // 20,480,000

typedef __attribute__((ext_vector_type(16))) __bf16 v16bf;
typedef __attribute__((ext_vector_type(8)))  __bf16 v8bf;
typedef __attribute__((ext_vector_type(8)))  float  v8f;

__device__ __forceinline__ void atomAddF(float* p, float v) {
  __hip_atomic_fetch_add(p, v, __ATOMIC_RELAXED, __HIP_MEMORY_SCOPE_AGENT);
}

// ---------------- elementwise / graph kernels ------------------------------

__global__ void k_zero_f4(float4* __restrict__ p, size_t n4) {
  size_t i = (size_t)blockIdx.x * blockDim.x + threadIdx.x;
  if (i < n4) p[i] = make_float4(0.f, 0.f, 0.f, 0.f);
}

__global__ void k_degree(const int* __restrict__ src, const int* __restrict__ dst,
                         const float* __restrict__ ew, float* __restrict__ deg) {
  int e = blockIdx.x * blockDim.x + threadIdx.x;
  if (e >= N_EDGES) return;
  float w = (src[e] == dst[e]) ? 0.f : ew[e];
  atomAddF(&deg[src[e]], w);
}

__global__ void k_norm(const int* __restrict__ src, const int* __restrict__ dst,
                       const float* __restrict__ ew, const float* __restrict__ deg,
                       float* __restrict__ nw) {
  int e = blockIdx.x * blockDim.x + threadIdx.x;
  if (e >= N_EDGES) return;
  int s = src[e], d = dst[e];
  float w  = (s == d) ? 0.f : ew[e];
  float ds = deg[s], dd = deg[d];
  float is = ds > 0.f ? rsqrtf(ds) : 0.f;
  float id = dd > 0.f ? rsqrtf(dd) : 0.f;
  nw[e] = -is * w * id;
}

__global__ void k_cvt_bf16(const float* __restrict__ in, __bf16* __restrict__ out, size_t n) {
  size_t i = (size_t)blockIdx.x * blockDim.x + threadIdx.x;
  if (i < n) out[i] = (__bf16)in[i];
}

// (L_hat T)[dst] += nw[e] * T[src]; one block (256 thr x 4 ch) per edge.
__global__ void k_lap_scatter(const int* __restrict__ src, const int* __restrict__ dst,
                              const float* __restrict__ nw, const __bf16* __restrict__ T,
                              float* __restrict__ acc) {
  int e = blockIdx.x;
  float w = nw[e];
  if (w == 0.f) return;
  size_t c0 = (size_t)threadIdx.x * 4;
  const __bf16* ps = T + (size_t)src[e] * CH + c0;
  float* pd = acc + (size_t)dst[e] * CH + c0;
#pragma unroll
  for (int i = 0; i < 4; ++i) atomAddF(pd + i, w * (float)ps[i]);
}

// Tnext = bf16(alpha*acc + beta*Tprev)
__global__ void k_combine(const float* __restrict__ acc, const __bf16* __restrict__ Tprev,
                          float alpha, float beta, __bf16* __restrict__ out, size_t n) {
  size_t i = (size_t)blockIdx.x * blockDim.x + threadIdx.x;
  if (i < n) out[i] = (__bf16)(alpha * acc[i] + beta * (float)Tprev[i]);
}

// W[k][c][j] f32  ->  Wt[k][j][c] bf16  (per weight tensor: 3 x 1024 x 1024)
__global__ void k_prep_w(const float* __restrict__ W, __bf16* __restrict__ Wt) {
  size_t idx = (size_t)blockIdx.x * 256 + threadIdx.x;   // 3*2^20 total
  size_t k = idx >> 20;
  size_t r = idx & 1048575u;
  size_t c = r >> 10;        // in-channel
  size_t j = r & 1023u;      // out-channel
  Wt[(k << 20) + (j << 10) + c] = (__bf16)W[idx];
}

// Z = sigmoid(Zpre) in place; R = sigmoid(Rpre); Tg0 = bf16(H * R)
__global__ void k_gate(float* __restrict__ Z, const float* __restrict__ Rpre,
                       const float* __restrict__ H, __bf16* __restrict__ Tg0, size_t n) {
  size_t i = (size_t)blockIdx.x * blockDim.x + threadIdx.x;
  if (i >= n) return;
  float z = 1.f / (1.f + expf(-Z[i]));
  float r = 1.f / (1.f + expf(-Rpre[i]));
  Z[i] = z;
  Tg0[i] = (__bf16)(H[i] * r);
}

// out = Z*H + (1-Z)*tanh(Hpre)
__global__ void k_final(const float* __restrict__ Z, const float* __restrict__ H,
                        const float* __restrict__ Hpre, float* __restrict__ out, size_t n) {
  size_t i = (size_t)blockIdx.x * blockDim.x + threadIdx.x;
  if (i >= n) return;
  float z = Z[i];
  out[i] = z * H[i] + (1.f - z) * tanhf(Hpre[i]);
}

// ---------------- WMMA GEMM: one GRU gate --------------------------------
// out[n,j] = sum_{g<2} sum_{k<3} A_g[k] (Nx1024 bf16, row-major) *
//            W_g[k] (bf16, pre-transposed to [j][c]) + b0[j] + b1[j]
// Wave tile: 16M x 128N (8 accumulator tiles). Block = 4 waves = 64M.
__global__ __launch_bounds__(128)
void k_gemm_gate(const __bf16* __restrict__ A0, const __bf16* __restrict__ W0,
                 const __bf16* __restrict__ A1, const __bf16* __restrict__ W1,
                 const float* __restrict__ b0, const float* __restrict__ b1,
                 float* __restrict__ out, int nrows) {
  const int lane = threadIdx.x & 31;
  const int wave = threadIdx.x >> 5;
  const int m0 = (blockIdx.y * 4 + wave) * 16;
  if (m0 >= nrows) return;                 // wave-uniform exit
  const int j0 = blockIdx.x * 128;

  const int nsel  = lane & 15;             // N column within 16 / A row within 16
  const int half  = lane >> 4;             // K-half selector

  v8f c[8] = {};                           // 8 x (16x16 f32) accumulators

  const __bf16* Ab[2] = {A0, A1};
  const __bf16* Wb[2] = {W0, W1};

  for (int g = 0; g < 2; ++g) {
    for (int k3 = 0; k3 < 3; ++k3) {
      const __bf16* A = Ab[g] + (size_t)k3 * (size_t)nrows * CH;
      const __bf16* W = Wb[g] + (size_t)k3 * CH * CH;
      // A lane base: row m0+nsel, K offset half*16 (contiguous K, matches B)
      const __bf16* arow = A + (size_t)(m0 + nsel) * CH + half * 16;
      // B lane base: Wt row j0+nsel, K offset half*16 (B 32x16 layout)
      const __bf16* wrow = W + (size_t)(j0 + nsel) * CH + half * 16;
      for (int kk = 0; kk < CH; kk += 32) {
        v8bf lo = *(const v8bf*)(arow + kk);        // K: half*16  .. +7
        v8bf hi = *(const v8bf*)(arow + kk + 8);    // K: half*16+8.. +7
        v16bf a = __builtin_shufflevector(lo, hi, 0, 1, 2, 3, 4, 5, 6, 7,
                                                  8, 9, 10, 11, 12, 13, 14, 15);
#pragma unroll
        for (int t = 0; t < 8; ++t) {
          v16bf b = *(const v16bf*)(wrow + (size_t)t * 16 * CH + kk);
          c[t] = __builtin_amdgcn_wmma_f32_16x16x32_bf16(
              false, a, false, b, (short)0, c[t], false, false);
        }
      }
    }
  }

  // Store per ISA f32 C/D layout: VGPR r -> M = (lane<16 ? r : 8+r), N = lane%16
  const int mh = half * 8;
#pragma unroll
  for (int t = 0; t < 8; ++t) {
    const int j = j0 + t * 16 + nsel;
    const float bias = b0[j] + b1[j];
#pragma unroll
    for (int r = 0; r < 8; ++r) {
      out[(size_t)(m0 + mh + r) * CH + j] = c[t][r] + bias;
    }
  }
}

// ---------------------------------------------------------------------------

extern "C" void kernel_launch(void* const* d_in, const int* in_sizes, int n_in,
                              void* d_out, int out_size, void* d_ws, size_t ws_size,
                              hipStream_t stream) {
  (void)in_sizes; (void)n_in; (void)out_size; (void)ws_size;

  const float* x  = (const float*)d_in[0];
  const float* H  = (const float*)d_in[1];
  const int*   ei = (const int*)d_in[2];
  const float* ew = (const float*)d_in[3];
  const float* Wsrc[6] = {(const float*)d_in[4],  (const float*)d_in[6],
                          (const float*)d_in[8],  (const float*)d_in[10],
                          (const float*)d_in[12], (const float*)d_in[14]};
  const float* bsrc[6] = {(const float*)d_in[5],  (const float*)d_in[7],
                          (const float*)d_in[9],  (const float*)d_in[11],
                          (const float*)d_in[13], (const float*)d_in[15]};
  // gate order: 0=Wxz 1=Whz 2=Wxr 3=Whr 4=Wxh 5=Whh
  const int* src = ei;
  const int* dst = ei + N_EDGES;
  float* out = (float*)d_out;

  // ---- workspace carve (all 256B aligned) ----
  char* p = (char*)d_ws;
  auto take = [&](size_t bytes) -> void* {
    void* r = (void*)p;
    p += (bytes + 255) & ~(size_t)255;
    return r;
  };
  __bf16* WT[6];
  for (int i = 0; i < 6; ++i) WT[i] = (__bf16*)take((size_t)3 * 1048576 * sizeof(__bf16));
  __bf16* Tx  = (__bf16*)take(3 * NCH * sizeof(__bf16));   // Tx0|Tx1|Tx2
  __bf16* Th  = (__bf16*)take(3 * NCH * sizeof(__bf16));
  __bf16* Tg  = (__bf16*)take(3 * NCH * sizeof(__bf16));
  float*  acc = (float*)take(NCH * sizeof(float));
  float*  Zb  = (float*)take(NCH * sizeof(float));         // Zpre -> Z
  float*  Rb  = (float*)take(NCH * sizeof(float));         // Rpre -> Hpre (reused)
  float*  deg = (float*)take(N_NODES * sizeof(float));
  float*  nw  = (float*)take(N_EDGES * sizeof(float));

  const int T256 = 256;
  const int eBlocks = (N_EDGES + T256 - 1) / T256;
  const size_t nElems = NCH;
  const int nBlocks = (int)((nElems + T256 - 1) / T256);
  const dim3 gemmGrid(CH / 128, (N_NODES / 16 + 3) / 4);   // (8, 313)

  // 1) weight transpose + bf16 convert
  for (int i = 0; i < 6; ++i)
    k_prep_w<<<3 * 1048576 / 256, 256, 0, stream>>>(Wsrc[i], WT[i]);

  // 2) graph normalization
  k_zero_f4<<<(N_NODES / 4 + T256 - 1) / T256, T256, 0, stream>>>((float4*)deg, N_NODES / 4);
  k_degree<<<eBlocks, T256, 0, stream>>>(src, dst, ew, deg);
  k_norm<<<eBlocks, T256, 0, stream>>>(src, dst, ew, deg, nw);

  // 3) base conversions
  k_cvt_bf16<<<nBlocks, T256, 0, stream>>>(x, Tx, nElems);
  k_cvt_bf16<<<nBlocks, T256, 0, stream>>>(H, Th, nElems);

  // 4) Chebyshev bases for x and H: T1 = L*T0 ; T2 = 2*L*T1 - T0
  __bf16* bases[2] = {Tx, Th};
  for (int bidx = 0; bidx < 2; ++bidx) {
    __bf16* T0 = bases[bidx];
    __bf16* T1 = T0 + NCH;
    __bf16* T2 = T0 + 2 * NCH;
    k_zero_f4<<<(int)(nElems / 4 + T256 - 1) / T256, T256, 0, stream>>>((float4*)acc, nElems / 4);
    k_lap_scatter<<<N_EDGES, T256, 0, stream>>>(src, dst, nw, T0, acc);
    k_combine<<<nBlocks, T256, 0, stream>>>(acc, T0, 1.f, 0.f, T1, nElems);
    k_zero_f4<<<(int)(nElems / 4 + T256 - 1) / T256, T256, 0, stream>>>((float4*)acc, nElems / 4);
    k_lap_scatter<<<N_EDGES, T256, 0, stream>>>(src, dst, nw, T1, acc);
    k_combine<<<nBlocks, T256, 0, stream>>>(acc, T0, 2.f, -1.f, T2, nElems);
  }

  // 5) Z and R pre-activations (6 GEMMs each, fused per gate)
  k_gemm_gate<<<gemmGrid, 128, 0, stream>>>(Tx, WT[0], Th, WT[1], bsrc[0], bsrc[1], Zb, N_NODES);
  k_gemm_gate<<<gemmGrid, 128, 0, stream>>>(Tx, WT[2], Th, WT[3], bsrc[2], bsrc[3], Rb, N_NODES);

  // 6) Z = sigmoid, R = sigmoid, Tg0 = bf16(H * R)
  k_gate<<<nBlocks, T256, 0, stream>>>(Zb, Rb, H, Tg, nElems);

  // 7) Chebyshev bases for H*R
  {
    __bf16* T0 = Tg;
    __bf16* T1 = Tg + NCH;
    __bf16* T2 = Tg + 2 * NCH;
    k_zero_f4<<<(int)(nElems / 4 + T256 - 1) / T256, T256, 0, stream>>>((float4*)acc, nElems / 4);
    k_lap_scatter<<<N_EDGES, T256, 0, stream>>>(src, dst, nw, T0, acc);
    k_combine<<<nBlocks, T256, 0, stream>>>(acc, T0, 1.f, 0.f, T1, nElems);
    k_zero_f4<<<(int)(nElems / 4 + T256 - 1) / T256, T256, 0, stream>>>((float4*)acc, nElems / 4);
    k_lap_scatter<<<N_EDGES, T256, 0, stream>>>(src, dst, nw, T1, acc);
    k_combine<<<nBlocks, T256, 0, stream>>>(acc, T0, 2.f, -1.f, T2, nElems);
  }

  // 8) candidate pre-activation (reuse Rb as Hpre)
  k_gemm_gate<<<gemmGrid, 128, 0, stream>>>(Tx, WT[4], Tg, WT[5], bsrc[4], bsrc[5], Rb, N_NODES);

  // 9) out = Z*H + (1-Z)*tanh(Hpre)
  k_final<<<nBlocks, T256, 0, stream>>>(Zb, H, Rb, out, nElems);
}
